// TitanMini_56607668961552
// MI455X (gfx1250) — matrix operand order repaired and checked by reading, not compiled, not run on hardware
//
#include <hip/hip_runtime.h>
#include <hip/hip_bf16.h>

typedef __attribute__((ext_vector_type(16))) _Float16 v16h;
typedef __attribute__((ext_vector_type(8)))  _Float16 v8h;
typedef __attribute__((ext_vector_type(8)))  float    v8f;

#define BATCH 128
#define SEQ   65
#define DMODEL 512
#define NHEAD 8
#define DKH   64
#define NLAYER 13
#define DFF   1920
#define MROWS (BATCH*SEQ)   // 8320 = 260*32, tiles divide exactly everywhere

// ---------------------------------------------------------------------------
// Weight prep: W [K x N] f32 row-major  ->  Wt [N x K] f16 row-major
// ---------------------------------------------------------------------------
__global__ void transpose_f16_kernel(const float* __restrict__ W,
                                     _Float16* __restrict__ Wt, int K, int N) {
    int idx = blockIdx.x * blockDim.x + threadIdx.x;
    if (idx >= K * N) return;
    int n = idx / K, k = idx - n * K;
    Wt[idx] = (_Float16)W[(size_t)k * N + n];
}

__global__ void pack_qkv_bias_kernel(const float* __restrict__ bq,
                                     const float* __restrict__ bk,
                                     const float* __restrict__ bv,
                                     float* __restrict__ bqkv) {
    int j = blockIdx.x * blockDim.x + threadIdx.x;
    if (j >= 3 * DMODEL) return;
    bqkv[j] = (j < DMODEL) ? bq[j] : (j < 2 * DMODEL) ? bk[j - DMODEL] : bv[j - 2 * DMODEL];
}

// ---------------------------------------------------------------------------
// WMMA GEMM: C[M x N](f32) = A[M x K](f16) * Bt^T(f16) + bias (+ Res)
// Bt is [N x K] f16. One wave computes a 32x64 tile (2 A frags x 4 B frags).
// blockDim = 256 (8 waves). A/B fragments use contiguous K order (element s = K 16*hi + s).
// ---------------------------------------------------------------------------
__global__ __launch_bounds__(256)
void gemm_wmma_kernel(const _Float16* __restrict__ A, const _Float16* __restrict__ Bt,
                      const float* __restrict__ bias, const float* __restrict__ Res,
                      float* __restrict__ C, int M, int N, int K) {
    const int wave   = blockIdx.x * 8 + (threadIdx.x >> 5);
    const int lane   = threadIdx.x & 31;
    const int tilesN = N >> 6;
    const int m0 = (wave / tilesN) << 5;     // 32 output rows per wave
    const int n0 = (wave % tilesN) << 6;     // 64 output cols per wave
    const int laneM = lane & 15;
    const int hi    = lane >> 4;             // 0: lanes 0-15, 1: lanes 16-31
    const int koffA = hi ? 16 : 0;           // A: hi lanes hold K 16-31
    const int koffB = hi ? 16 : 0;           // B: hi lanes hold K 16-31

    v8f acc[8] = {};                         // [mi*4 + nt]
    const _Float16* a0p = A  + (size_t)(m0 + laneM) * K + koffA;
    const _Float16* a1p = A  + (size_t)(m0 + 16 + laneM) * K + koffA;
    const _Float16* bp  = Bt + (size_t)(n0 + laneM) * K + koffB;

    for (int kt = 0; kt < K; kt += 32) {
        // A fragments: halves 0-7 = K kt+koffA.., halves 8-15 = K kt+8+koffA..
        v8h al0 = *(const v8h*)(a0p + kt);
        v8h ah0 = *(const v8h*)(a0p + kt + 8);
        v8h al1 = *(const v8h*)(a1p + kt);
        v8h ah1 = *(const v8h*)(a1p + kt + 8);
        v16h a0 = __builtin_shufflevector(al0, ah0, 0,1,2,3,4,5,6,7,8,9,10,11,12,13,14,15);
        v16h a1 = __builtin_shufflevector(al1, ah1, 0,1,2,3,4,5,6,7,8,9,10,11,12,13,14,15);
#pragma unroll
        for (int nt = 0; nt < 4; ++nt) {
            v16h b = *(const v16h*)(bp + (size_t)(nt * 16) * K + kt);
            acc[nt]     = __builtin_amdgcn_wmma_f32_16x16x32_f16(false, a0, false, b, (short)0, acc[nt],     false, false);
            acc[4 + nt] = __builtin_amdgcn_wmma_f32_16x16x32_f16(false, a1, false, b, (short)0, acc[4 + nt], false, false);
        }
    }

    // Epilogue: D layout — VGPR r holds (M = tileRow + r + 8*hi, N = n0 + nt*16 + laneM)
    const int ncol = n0 + laneM;
#pragma unroll
    for (int mi = 0; mi < 2; ++mi) {
#pragma unroll
        for (int nt = 0; nt < 4; ++nt) {
            int n = ncol + nt * 16;
            float bia = bias[n];
#pragma unroll
            for (int r = 0; r < 8; ++r) {
                int m = m0 + mi * 16 + r + (hi ? 8 : 0);
                size_t idx = (size_t)m * N + n;
                float v = acc[mi * 4 + nt][r] + bia;
                if (Res) v += Res[idx];
                C[idx] = v;
            }
        }
    }
}

// ---------------------------------------------------------------------------
// Embedding: h[b,0,:] = cls; h[b,1+s,:] = x[b,:,s] @ conv_w + conv_b + pos(s)
// Residual stream h stays f32.
// ---------------------------------------------------------------------------
__global__ void embed_kernel(const float* __restrict__ x, const float* __restrict__ conv_w,
                             const float* __restrict__ conv_b, const float* __restrict__ cls_token,
                             const float* __restrict__ abs_pos, const float* __restrict__ file_emb,
                             const float* __restrict__ rank_emb, const float* __restrict__ diag_emb,
                             const float* __restrict__ adiag_emb, float* __restrict__ h) {
    int row = blockIdx.x;                 // 0 .. B*65-1
    int b = row / SEQ, s65 = row - b * SEQ;
    int tid = threadIdx.x;                // 256
    float* out = h + (size_t)row * DMODEL;
    if (s65 == 0) {
        out[tid] = cls_token[tid];
        out[tid + 256] = cls_token[tid + 256];
        return;
    }
    int s = s65 - 1;
    int rk = s >> 3, fl = s & 7;
    __shared__ float xs[112];
    if (tid < 112) xs[tid] = x[((size_t)b * 112 + tid) * 64 + s];
    __syncthreads();
#pragma unroll
    for (int dd = 0; dd < 2; ++dd) {
        int d = tid + dd * 256;
        float acc = conv_b[d] + abs_pos[s * DMODEL + d] + file_emb[fl * DMODEL + d]
                  + rank_emb[rk * DMODEL + d] + diag_emb[(rk + fl) * DMODEL + d]
                  + adiag_emb[(rk - fl + 7) * DMODEL + d];
        for (int c = 0; c < 112; ++c) acc += xs[c] * conv_w[c * DMODEL + d];
        out[d] = acc;
    }
}

// ---------------------------------------------------------------------------
// LayerNorm over D=512, one block (256 threads) per row; f32 in, f16 out
// (output feeds WMMA GEMMs / heads only)
// ---------------------------------------------------------------------------
__global__ void layernorm_kernel(const float* __restrict__ X, const float* __restrict__ g,
                                 const float* __restrict__ bta, _Float16* __restrict__ Y) {
    __shared__ float red[256];
    int row = blockIdx.x, tid = threadIdx.x;
    const float* xr = X + (size_t)row * DMODEL;
    float x0 = xr[tid], x1 = xr[tid + 256];
    red[tid] = x0 + x1;
    __syncthreads();
    for (int s = 128; s > 0; s >>= 1) { if (tid < s) red[tid] += red[tid + s]; __syncthreads(); }
    float mean = red[0] * (1.0f / DMODEL);
    __syncthreads();
    float d0 = x0 - mean, d1 = x1 - mean;
    red[tid] = d0 * d0 + d1 * d1;
    __syncthreads();
    for (int s = 128; s > 0; s >>= 1) { if (tid < s) red[tid] += red[tid + s]; __syncthreads(); }
    float rstd = rsqrtf(red[0] * (1.0f / DMODEL) + 1e-5f);
    Y[(size_t)row * DMODEL + tid]       = (_Float16)(d0 * rstd * g[tid] + bta[tid]);
    Y[(size_t)row * DMODEL + tid + 256] = (_Float16)(d1 * rstd * g[tid + 256] + bta[tid + 256]);
}

// ---------------------------------------------------------------------------
// Attention: one block per (b, head). qkv: [M x 1536] f32 (q|k|v).
// ctx: [M x 512] f16 (feeds the O-proj WMMA GEMM).
// ---------------------------------------------------------------------------
__global__ void attention_kernel(const float* __restrict__ qkv,
                                 const float* __restrict__ rel_bias,
                                 _Float16* __restrict__ ctx, int layer) {
    __shared__ float qs[SEQ * DKH];
    __shared__ float ks[SEQ * DKH];
    __shared__ float sc[SEQ * SEQ];
    int bh = blockIdx.x;
    int b = bh >> 3, hd = bh & 7;
    int tid = threadIdx.x;                // 128
    const float scale = 0.125f;           // 1/sqrt(64)
    const float* qbase = qkv + (size_t)b * SEQ * (3 * DMODEL) + hd * DKH;

    for (int idx = tid; idx < SEQ * DKH; idx += 128) {
        int s = idx >> 6, d = idx & 63;
        const float* rowp = qbase + (size_t)s * (3 * DMODEL);
        qs[idx] = rowp[d] * scale;
        ks[idx] = rowp[DMODEL + d];
    }
    __syncthreads();

    const float* rb = rel_bias + (size_t)layer * 15 * 15 * NHEAD + hd;
    for (int idx = tid; idx < SEQ * SEQ; idx += 128) {
        int i = idx / SEQ, j = idx - i * SEQ;
        float acc = 0.f;
        const float* qi = qs + i * DKH;
        const float* kj = ks + j * DKH;
#pragma unroll 8
        for (int d = 0; d < DKH; ++d) acc += qi[d] * kj[d];
        if (i > 0 && j > 0) {
            int si = i - 1, sj = j - 1;
            int ri = si >> 3, fi = si & 7, rj = sj >> 3, fj = sj & 7;
            int dr = ri - rj; dr = dr < -7 ? -7 : (dr > 7 ? 7 : dr);
            int df = fi - fj; df = df < -7 ? -7 : (df > 7 ? 7 : df);
            acc += rb[((dr + 7) * 15 + (df + 7)) * NHEAD];
        }
        sc[idx] = acc;
    }
    __syncthreads();

    if (tid < SEQ) {                      // softmax per query row
        float* r = sc + tid * SEQ;
        float mx = r[0];
        for (int j = 1; j < SEQ; ++j) mx = fmaxf(mx, r[j]);
        float sum = 0.f;
        for (int j = 0; j < SEQ; ++j) { float e = __expf(r[j] - mx); r[j] = e; sum += e; }
        float inv = 1.f / sum;
        for (int j = 0; j < SEQ; ++j) r[j] *= inv;
    }
    __syncthreads();

    for (int idx = tid; idx < SEQ * DKH; idx += 128) {  // reuse qs for V
        int s = idx >> 6, d = idx & 63;
        qs[idx] = qbase[(size_t)s * (3 * DMODEL) + 2 * DMODEL + d];
    }
    __syncthreads();

    for (int idx = tid; idx < SEQ * DKH; idx += 128) {
        int i = idx >> 6, d = idx & 63;
        float acc = 0.f;
        const float* ar = sc + i * SEQ;
        for (int j = 0; j < SEQ; ++j) acc += ar[j] * qs[j * DKH + d];
        ctx[((size_t)b * SEQ + i) * DMODEL + hd * DKH + d] = (_Float16)acc;
    }
}

// ---------------------------------------------------------------------------
// GEGLU: G[m,j] = gelu(U[m,j]) * U[m,1920+j]; f32 in, f16 out (feeds FF2 GEMM)
// ---------------------------------------------------------------------------
__global__ void geglu_kernel(const float* __restrict__ U, _Float16* __restrict__ G, int total) {
    int idx = blockIdx.x * blockDim.x + threadIdx.x;
    if (idx >= total) return;
    int m = idx / DFF, j = idx - m * DFF;
    float a = U[(size_t)m * (2 * DFF) + j];
    float b = U[(size_t)m * (2 * DFF) + DFF + j];
    float ge = 0.5f * a * (1.f + erff(a * 0.70710678118654752f));
    G[idx] = (_Float16)(ge * b);
}

// ---------------------------------------------------------------------------
// Heads (read the f16 final-LN output, compute in f32)
// ---------------------------------------------------------------------------
__global__ void value_head_kernel(const _Float16* __restrict__ hn,
                                  const float* __restrict__ vw1, const float* __restrict__ vb1,
                                  const float* __restrict__ vw2, const float* __restrict__ vb2,
                                  const float* __restrict__ vw3, const float* __restrict__ vb3,
                                  float* __restrict__ out_value) {
    __shared__ float cls[DMODEL];
    __shared__ float v1s[256];
    __shared__ float v2s[128];
    int b = blockIdx.x, tid = threadIdx.x;   // 256
    const _Float16* src = hn + (size_t)b * SEQ * DMODEL;
    cls[tid] = (float)src[tid]; cls[tid + 256] = (float)src[tid + 256];
    __syncthreads();
    {
        float acc = vb1[tid];
        for (int k = 0; k < DMODEL; ++k) acc += cls[k] * vw1[k * 256 + tid];
        v1s[tid] = 0.5f * acc * (1.f + erff(acc * 0.70710678118654752f));
    }
    __syncthreads();
    if (tid < 128) {
        float acc = vb2[tid];
        for (int k = 0; k < 256; ++k) acc += v1s[k] * vw2[k * 128 + tid];
        v2s[tid] = 0.5f * acc * (1.f + erff(acc * 0.70710678118654752f));
    }
    __syncthreads();
    if (tid < 3) {
        float acc = vb3[tid];
        for (int k = 0; k < 128; ++k) acc += v2s[k] * vw3[k * 3 + tid];
        out_value[b * 3 + tid] = acc;
    }
}

__global__ void policy_head_kernel(const _Float16* __restrict__ hn, const float* __restrict__ pw,
                                   const float* __restrict__ pb, float* __restrict__ out_policy) {
    __shared__ float rowv[DMODEL];
    int blk = blockIdx.x;                 // 0..8191
    int b = blk >> 6, s = blk & 63;
    int tid = threadIdx.x;                // 128
    const _Float16* src = hn + ((size_t)b * SEQ + 1 + s) * DMODEL;
    for (int i = tid; i < DMODEL; i += 128) rowv[i] = (float)src[i];
    __syncthreads();
    if (tid < 72) {
        float acc = pb[tid];
        for (int k = 0; k < DMODEL; ++k) acc += rowv[k] * pw[k * 72 + tid];
        out_policy[((size_t)b * 64 + s) * 72 + tid] = acc;
    }
}

// ---------------------------------------------------------------------------
// Host
// ---------------------------------------------------------------------------
extern "C" void kernel_launch(void* const* d_in, const int* in_sizes, int n_in,
                              void* d_out, int out_size, void* d_ws, size_t ws_size,
                              hipStream_t stream) {
    const float* x          = (const float*)d_in[0];
    const float* conv_w     = (const float*)d_in[1];
    const float* conv_b     = (const float*)d_in[2];
    const float* cls_token  = (const float*)d_in[3];
    const float* abs_pos    = (const float*)d_in[4];
    const float* file_emb   = (const float*)d_in[5];
    const float* rank_emb   = (const float*)d_in[6];
    const float* diag_emb   = (const float*)d_in[7];
    const float* adiag_emb  = (const float*)d_in[8];
    const float* ln1_g      = (const float*)d_in[9];
    const float* ln1_b      = (const float*)d_in[10];
    const float* wq         = (const float*)d_in[11];
    const float* bq         = (const float*)d_in[12];
    const float* wk         = (const float*)d_in[13];
    const float* bk         = (const float*)d_in[14];
    const float* wv         = (const float*)d_in[15];
    const float* bv         = (const float*)d_in[16];
    const float* wo         = (const float*)d_in[17];
    const float* bo         = (const float*)d_in[18];
    const float* rel_bias   = (const float*)d_in[19];
    const float* ln2_g      = (const float*)d_in[20];
    const float* ln2_b      = (const float*)d_in[21];
    const float* w_ff1      = (const float*)d_in[22];
    const float* b_ff1      = (const float*)d_in[23];
    const float* w_ff2      = (const float*)d_in[24];
    const float* b_ff2      = (const float*)d_in[25];
    const float* out_g      = (const float*)d_in[26];
    const float* out_b      = (const float*)d_in[27];
    const float* vw1        = (const float*)d_in[28];
    const float* vb1        = (const float*)d_in[29];
    const float* vw2        = (const float*)d_in[30];
    const float* vb2        = (const float*)d_in[31];
    const float* vw3        = (const float*)d_in[32];
    const float* vb3        = (const float*)d_in[33];
    const float* pw         = (const float*)d_in[34];
    const float* pb         = (const float*)d_in[35];

    char* ws = (char*)d_ws;
    size_t off = 0;
    auto alloc = [&](size_t bytes) -> void* {
        off = (off + 255) & ~(size_t)255;
        void* p = ws + off;
        off += bytes;
        return p;
    };

    float*    hbuf   = (float*)alloc((size_t)MROWS * DMODEL * 4);          // residual (f32)
    _Float16* nbuf   = (_Float16*)alloc((size_t)MROWS * DMODEL * 2);       // LN out (f16)
    float*    qkvbuf = (float*)alloc((size_t)MROWS * 3 * DMODEL * 4);      // q|k|v (f32)
    _Float16* ctxbuf = (_Float16*)alloc((size_t)MROWS * DMODEL * 2);       // attn ctx (f16)
    float*    ubuf   = (float*)alloc((size_t)MROWS * 2 * DFF * 4);         // FF1 out (f32)
    _Float16* gbuf   = (_Float16*)alloc((size_t)MROWS * DFF * 2);          // GEGLU out (f16)
    _Float16* wqkv_t = (_Float16*)alloc((size_t)NLAYER * 3 * DMODEL * DMODEL * 2);
    float*    bqkv   = (float*)alloc((size_t)NLAYER * 3 * DMODEL * 4);
    _Float16* wo_t   = (_Float16*)alloc((size_t)NLAYER * DMODEL * DMODEL * 2);
    _Float16* wff1_t = (_Float16*)alloc((size_t)NLAYER * 2 * DFF * DMODEL * 2);
    _Float16* wff2_t = (_Float16*)alloc((size_t)NLAYER * DMODEL * DFF * 2);

    // ---- weight prep (transpose + f32->f16, pack QKV) ----
    const int TE = 256;
    for (int l = 0; l < NLAYER; ++l) {
        _Float16* dstq = wqkv_t + (size_t)l * 3 * DMODEL * DMODEL;
        transpose_f16_kernel<<<(DMODEL * DMODEL + TE - 1) / TE, TE, 0, stream>>>(
            wq + (size_t)l * DMODEL * DMODEL, dstq, DMODEL, DMODEL);
        transpose_f16_kernel<<<(DMODEL * DMODEL + TE - 1) / TE, TE, 0, stream>>>(
            wk + (size_t)l * DMODEL * DMODEL, dstq + (size_t)DMODEL * DMODEL, DMODEL, DMODEL);
        transpose_f16_kernel<<<(DMODEL * DMODEL + TE - 1) / TE, TE, 0, stream>>>(
            wv + (size_t)l * DMODEL * DMODEL, dstq + (size_t)2 * DMODEL * DMODEL, DMODEL, DMODEL);
        transpose_f16_kernel<<<(DMODEL * DMODEL + TE - 1) / TE, TE, 0, stream>>>(
            wo + (size_t)l * DMODEL * DMODEL, wo_t + (size_t)l * DMODEL * DMODEL, DMODEL, DMODEL);
        transpose_f16_kernel<<<(DMODEL * 2 * DFF + TE - 1) / TE, TE, 0, stream>>>(
            w_ff1 + (size_t)l * DMODEL * 2 * DFF, wff1_t + (size_t)l * 2 * DFF * DMODEL, DMODEL, 2 * DFF);
        transpose_f16_kernel<<<(DFF * DMODEL + TE - 1) / TE, TE, 0, stream>>>(
            w_ff2 + (size_t)l * DFF * DMODEL, wff2_t + (size_t)l * DMODEL * DFF, DFF, DMODEL);
        pack_qkv_bias_kernel<<<(3 * DMODEL + TE - 1) / TE, TE, 0, stream>>>(
            bq + (size_t)l * DMODEL, bk + (size_t)l * DMODEL, bv + (size_t)l * DMODEL,
            bqkv + (size_t)l * 3 * DMODEL);
    }

    auto gemm = [&](const _Float16* A, const _Float16* Bt, const float* bias, const float* Res,
                    float* C, int M, int N, int K) {
        int waves = (M / 32) * (N / 64);
        gemm_wmma_kernel<<<waves / 8, 256, 0, stream>>>(A, Bt, bias, Res, C, M, N, K);
    };

    // ---- embedding ----
    embed_kernel<<<MROWS, 256, 0, stream>>>(x, conv_w, conv_b, cls_token, abs_pos,
                                            file_emb, rank_emb, diag_emb, adiag_emb, hbuf);

    // ---- transformer blocks ----
    for (int l = 0; l < NLAYER; ++l) {
        layernorm_kernel<<<MROWS, 256, 0, stream>>>(hbuf, ln1_g + (size_t)l * DMODEL,
                                                    ln1_b + (size_t)l * DMODEL, nbuf);
        gemm(nbuf, wqkv_t + (size_t)l * 3 * DMODEL * DMODEL,
             bqkv + (size_t)l * 3 * DMODEL, nullptr, qkvbuf, MROWS, 3 * DMODEL, DMODEL);
        attention_kernel<<<BATCH * NHEAD, 128, 0, stream>>>(qkvbuf, rel_bias, ctxbuf, l);
        gemm(ctxbuf, wo_t + (size_t)l * DMODEL * DMODEL,
             bo + (size_t)l * DMODEL, hbuf, hbuf, MROWS, DMODEL, DMODEL);
        layernorm_kernel<<<MROWS, 256, 0, stream>>>(hbuf, ln2_g + (size_t)l * DMODEL,
                                                    ln2_b + (size_t)l * DMODEL, nbuf);
        gemm(nbuf, wff1_t + (size_t)l * 2 * DFF * DMODEL,
             b_ff1 + (size_t)l * 2 * DFF, nullptr, ubuf, MROWS, 2 * DFF, DMODEL);
        int gel = MROWS * DFF;
        geglu_kernel<<<(gel + TE - 1) / TE, TE, 0, stream>>>(ubuf, gbuf, gel);
        gemm(gbuf, wff2_t + (size_t)l * DMODEL * DFF,
             b_ff2 + (size_t)l * DMODEL, hbuf, hbuf, MROWS, DMODEL, DFF);
    }

    // ---- final norm + heads ----
    layernorm_kernel<<<MROWS, 256, 0, stream>>>(hbuf, out_g, out_b, nbuf);
    float* out = (float*)d_out;
    policy_head_kernel<<<BATCH * 64, 128, 0, stream>>>(nbuf, pw, pb, out);
    value_head_kernel<<<BATCH, 256, 0, stream>>>(nbuf, vw1, vb1, vw2, vb2, vw3, vb3,
                                                 out + (size_t)BATCH * 64 * 72);
}
